// DiffSeg_13065290515040
// MI455X (gfx1250) — compile-verified, matrix operand order checked by reading, not run on hardware
//
#include <hip/hip_runtime.h>
#include <hip/hip_bf16.h>
#include <math.h>

typedef __attribute__((ext_vector_type(16))) _Float16 v16h;
typedef __attribute__((ext_vector_type(8)))  _Float16 v8h;
typedef __attribute__((ext_vector_type(8)))  float    v8f;

#define KL_THR 0.9f

__device__ __forceinline__ int anchor_row(int a) {
    // grid = linspace(1,61,16) in both dims; row = i*64 + j
    return (1 + 4 * (a >> 4)) * 64 + (1 + 4 * (a & 15));
}

// ---------------------------------------------------------------------------
// Kernel 1: one block per query pixel m (4096 blocks x 256 threads).
// Computes the aggregated attention row agg[m,:] (4096 values) on the fly and
// directly emits Yh = fp16(agg), lY = fp16(log(fp16(agg))) and the per-row
// entropy sy[m] = sum(Yh*lY) (fp32 accum). agg itself is never materialized.
// ---------------------------------------------------------------------------
__global__ __launch_bounds__(256)
void k_aggregate(const float* __restrict__ w64, const float* __restrict__ w32,
                 const float* __restrict__ w16, const float* __restrict__ w8,
                 _Float16* __restrict__ Yh, _Float16* __restrict__ lY,
                 float* __restrict__ sy)
{
    __shared__ float sB[4096];
    __shared__ float sRed[256];
    const int m = blockIdx.x;
    const int t = threadIdx.x;
    const int qi = m >> 6, qj = m & 63;

    float acc[16];
#pragma unroll
    for (int k = 0; k < 16; ++k) acc[k] = 0.f;

    const float* srcs[4] = {w64, w32, w16, w8};
    const int ns[4] = {64, 32, 16, 8};

    for (int lev = 0; lev < 4; ++lev) {
        const int n = ns[lev];
        const int n2 = n * n;
        const float wr = (float)n / 120.f;          // size ratio
        const int qrow = (qi % n) * n + (qj % n);   // jnp.tile indexing
        const float* wsrc = srcs[lev];
        // head-sum base map into LDS (mean scale cancels with normalization)
        for (int idx = t; idx < n2; idx += 256) {
            float s = 0.f;
            for (int h = 0; h < 8; ++h)
                s += wsrc[((size_t)h * n2 + qrow) * (size_t)n2 + idx];
            sB[idx] = s;
        }
        __syncthreads();

        float upv[16];
        float psum = 0.f;
        if (n == 64) {
#pragma unroll
            for (int k = 0; k < 16; ++k) {
                int p = t + 256 * k;
                float v = sB[p];
                upv[k] = v; psum += v;
            }
        } else {
            const float sc = (float)(n - 1) / 63.f;   // linspace(0, n-1, 64)
#pragma unroll
            for (int k = 0; k < 16; ++k) {
                int p = t + 256 * k;
                int y = p >> 6, x = p & 63;
                float fy = sc * (float)y;
                float fx = sc * (float)x;
                int y0 = (int)fy; if (y0 > n - 2) y0 = n - 2;
                int x0 = (int)fx; if (x0 > n - 2) x0 = n - 2;
                float wy = fy - (float)y0, wx = fx - (float)x0;
                float a = sB[y0 * n + x0],       b = sB[y0 * n + x0 + 1];
                float c = sB[(y0 + 1) * n + x0], d = sB[(y0 + 1) * n + x0 + 1];
                float r0 = a + (c - a) * wy;   // interp along y first (ref order)
                float r1 = b + (d - b) * wy;
                float v = r0 + (r1 - r0) * wx;
                upv[k] = v; psum += v;
            }
        }
        // block-reduce map sum for normalization
        sRed[t] = psum;
        __syncthreads();
        for (int sft = 128; sft > 0; sft >>= 1) {
            if (t < sft) sRed[t] += sRed[t + sft];
            __syncthreads();
        }
        float inv = wr / sRed[0];
        __syncthreads();
#pragma unroll
        for (int k = 0; k < 16; ++k) acc[k] += upv[k] * inv;
        __syncthreads();   // sB reused next level
    }

    // emit fp16 Y row, fp16 log row, and entropy
    float esum = 0.f;
#pragma unroll
    for (int k = 0; k < 16; ++k) {
        int p = t + 256 * k;
        _Float16 h  = (_Float16)acc[k];
        _Float16 lh = (_Float16)logf((float)h);
        Yh[(size_t)m * 4096 + p] = h;
        lY[(size_t)m * 4096 + p] = lh;
        esum += (float)(_Float16)(h * lh);
    }
    sRed[t] = esum;
    __syncthreads();
    for (int sft = 128; sft > 0; sft >>= 1) {
        if (t < sft) sRed[t] += sRed[t + sft];
        __syncthreads();
    }
    if (t == 0) sy[m] = sRed[0];
}

// ---------------------------------------------------------------------------
// Kernel 2: cross = X*lY^T + lX*Yh^T (256x4096, K=4096) with
// v_wmma_f32_16x16x32_f16; directly thresholds KL into a binary fp16 matrix.
// One 16x16 tile per wave, 8 waves/block, 512 blocks.
// A-fragment layout (16-bit 16x32): lane L<16 -> row L, K {0..7,16..23};
// lane L>=16 -> row L-16, K {8..15,24..31}; B mirrors with N on lanes.
// ---------------------------------------------------------------------------
__global__ __launch_bounds__(256)
void k_cross_kl(const _Float16* __restrict__ Yh, const _Float16* __restrict__ lY,
                const float* __restrict__ sy, _Float16* __restrict__ klnum)
{
    const int lane = threadIdx.x & 31;
    const int wave = threadIdx.x >> 5;
    const int tile = blockIdx.x * 8 + wave;
    const int mt = tile >> 8;       // 0..15  (anchor tiles)
    const int nt = tile & 255;      // 0..255 (pixel tiles)
    const int mloc = lane & 15;
    const int sel  = lane >> 4;

    const size_t aoff = (size_t)anchor_row(mt * 16 + mloc) * 4096;
    const size_t boff = (size_t)(nt * 16 + mloc) * 4096;

    v8f c = {};
    for (int kb = 0; kb < 4096; kb += 32) {
        const _Float16* pa1 = Yh + aoff + kb + sel * 8;
        const _Float16* pa2 = lY + aoff + kb + sel * 8;
        const _Float16* pb1 = lY + boff + kb + sel * 8;
        const _Float16* pb2 = Yh + boff + kb + sel * 8;
        v8h a1lo = *(const v8h*)pa1, a1hi = *(const v8h*)(pa1 + 16);
        v8h a2lo = *(const v8h*)pa2, a2hi = *(const v8h*)(pa2 + 16);
        v8h b1lo = *(const v8h*)pb1, b1hi = *(const v8h*)(pb1 + 16);
        v8h b2lo = *(const v8h*)pb2, b2hi = *(const v8h*)(pb2 + 16);
        __builtin_prefetch(pa1 + 128, 0, 3);   // global_prefetch_b8
        __builtin_prefetch(pb1 + 128, 0, 3);
        v16h A1, A2, B1, B2;
#pragma unroll
        for (int i = 0; i < 8; ++i) {
            A1[i] = a1lo[i]; A1[i + 8] = a1hi[i];
            A2[i] = a2lo[i]; A2[i + 8] = a2hi[i];
            B1[i] = b1lo[i]; B1[i + 8] = b1hi[i];
            B2[i] = b2lo[i]; B2[i + 8] = b2hi[i];
        }
        c = __builtin_amdgcn_wmma_f32_16x16x32_f16(false, A1, false, B1,
                                                   (short)0, c, false, false);
        c = __builtin_amdgcn_wmma_f32_16x16x32_f16(false, A2, false, B2,
                                                   (short)0, c, false, false);
    }
    const int col = nt * 16 + mloc;
    const float syc = sy[col];
#pragma unroll
    for (int r = 0; r < 8; ++r) {
        int aidx = mt * 16 + r + sel * 8;
        float sx = sy[anchor_row(aidx)];           // anchors are rows of Yh
        float kl = 0.5f * (sx + syc - c[r]);
        klnum[(size_t)aidx * 4096 + col] = (kl < KL_THR) ? (_Float16)1.0f
                                                         : (_Float16)0.0f;
    }
}

// Row counts of the binary selection matrix.
__global__ __launch_bounds__(256)
void k_rowcount(const _Float16* __restrict__ klnum, float* __restrict__ cnt)
{
    __shared__ float sRed[256];
    const int a = blockIdx.x, t = threadIdx.x;
    float s = 0.f;
    for (int p = t; p < 4096; p += 256) s += (float)klnum[(size_t)a * 4096 + p];
    sRed[t] = s; __syncthreads();
    for (int sft = 128; sft > 0; sft >>= 1) {
        if (t < sft) sRed[t] += sRed[t + sft];
        __syncthreads();
    }
    if (t == 0) cnt[a] = sRed[0];
}

// fp16 4096x4096 transpose (LDS tiled) so the next GEMM's B loads stay contiguous.
__global__ __launch_bounds__(256)
void k_transpose_h(const _Float16* __restrict__ src, _Float16* __restrict__ dst)
{
    __shared__ _Float16 tile[32][33];
    const int bx = blockIdx.x, by = blockIdx.y;
    const int tx = threadIdx.x & 31, tq = threadIdx.x >> 5;
#pragma unroll
    for (int q = 0; q < 4; ++q) {
        int row = tq * 4 + q;
        tile[row][tx] = src[(size_t)(by * 32 + row) * 4096 + bx * 32 + tx];
    }
    __syncthreads();
#pragma unroll
    for (int q = 0; q < 4; ++q) {
        int row = tq * 4 + q;
        dst[(size_t)(bx * 32 + row) * 4096 + by * 32 + tx] = tile[tx][row];
    }
}

// ---------------------------------------------------------------------------
// Kernel 3: new = klnum @ Yh / cnt  (256x4096, K=4096) via WMMA.
// B[k][n] = Yh[k][col] = YhT[col][k] (contiguous per lane).
// ---------------------------------------------------------------------------
__global__ __launch_bounds__(256)
void k_newattns(const _Float16* __restrict__ klnum, const _Float16* __restrict__ YhT,
                const float* __restrict__ cnt, float* __restrict__ attns)
{
    const int lane = threadIdx.x & 31;
    const int wave = threadIdx.x >> 5;
    const int tile = blockIdx.x * 8 + wave;
    const int mt = tile >> 8;
    const int nt = tile & 255;
    const int mloc = lane & 15;
    const int sel  = lane >> 4;

    const size_t aoff = (size_t)(mt * 16 + mloc) * 4096;
    const size_t boff = (size_t)(nt * 16 + mloc) * 4096;

    v8f c = {};
    for (int kb = 0; kb < 4096; kb += 32) {
        const _Float16* pa = klnum + aoff + kb + sel * 8;
        const _Float16* pb = YhT   + boff + kb + sel * 8;
        v8h alo = *(const v8h*)pa, ahi = *(const v8h*)(pa + 16);
        v8h blo = *(const v8h*)pb, bhi = *(const v8h*)(pb + 16);
        __builtin_prefetch(pa + 128, 0, 3);
        __builtin_prefetch(pb + 128, 0, 3);
        v16h A, B;
#pragma unroll
        for (int i = 0; i < 8; ++i) {
            A[i] = alo[i]; A[i + 8] = ahi[i];
            B[i] = blo[i]; B[i + 8] = bhi[i];
        }
        c = __builtin_amdgcn_wmma_f32_16x16x32_f16(false, A, false, B,
                                                   (short)0, c, false, false);
    }
    const int col = nt * 16 + mloc;
#pragma unroll
    for (int r = 0; r < 8; ++r) {
        int aidx = mt * 16 + r + sel * 8;
        attns[(size_t)aidx * 4096 + col] = c[r] / cnt[aidx];
    }
}

__global__ void k_init_valid(int* valid) { valid[threadIdx.x] = 1; }

// Greedy-merge prep: lX = valid ? log(X) : 0 ; sx = sum(X*lX)  (fp32, ref exact)
__global__ __launch_bounds__(256)
void k_prep(const float* __restrict__ X, const int* __restrict__ valid,
            float* __restrict__ lX, float* __restrict__ sx)
{
    __shared__ float sRed[256];
    const int row = blockIdx.x, t = threadIdx.x;
    const int v = valid[row];
    float s = 0.f;
    for (int p = t; p < 4096; p += 256) {
        float x = X[(size_t)row * 4096 + p];
        float l = v ? logf(x) : 0.f;
        lX[(size_t)row * 4096 + p] = l;
        s += x * l;
    }
    sRed[t] = s; __syncthreads();
    for (int sft = 128; sft > 0; sft >>= 1) {
        if (t < sft) sRed[t] += sRed[t + sft];
        __syncthreads();
    }
    if (t == 0) sx[row] = sRed[0];
}

// 256x256 KL matrix (K=4096, fp32 as in reference), inf on invalid columns.
__global__ __launch_bounds__(256)
void k_klmat(const float* __restrict__ X, const float* __restrict__ lX,
             const float* __restrict__ sx, const int* __restrict__ valid,
             float* __restrict__ klmat)
{
    const int i = blockIdx.x, j = threadIdx.x;
    const float* Xi  = X  + (size_t)i * 4096;
    const float* lXi = lX + (size_t)i * 4096;
    const float* Xj  = X  + (size_t)j * 4096;
    const float* lXj = lX + (size_t)j * 4096;
    float acc = 0.f;
    for (int p = 0; p < 4096; ++p)
        acc += Xi[p] * lXj[p] + lXi[p] * Xj[p];
    float v = 0.5f * (sx[i] + sx[j] - acc);
    klmat[i * 256 + j] = valid[j] ? v : INFINITY;
}

// Serial greedy merge: single workgroup, LDS state, 256 sequential steps.
__global__ __launch_bounds__(256)
void k_greedy(const float* __restrict__ klmat, int* __restrict__ valid,
              float* __restrict__ sel, int* __restrict__ ocptr)
{
    __shared__ int matched[256];
    __shared__ int svalid[256];
    __shared__ int activeFlag;
    __shared__ int oc;
    const int t = threadIdx.x;
    matched[t] = 0;
    svalid[t] = valid[t];
    for (int rr = 0; rr < 256; ++rr) sel[rr * 256 + t] = 0.f;
    if (t == 0) oc = 0;
    __syncthreads();
    for (int i = 0; i < 256; ++i) {
        if (t == 0) activeFlag = (svalid[i] && !matched[i]) ? 1 : 0;
        __syncthreads();
        int row = (svalid[t] && (klmat[i * 256 + t] < KL_THR)) ? 1 : 0;
        if (activeFlag) {
            if (row) matched[t] = 1;
            sel[oc * 256 + t] = (float)row;
        }
        __syncthreads();
        if (t == 0 && activeFlag) oc++;
        __syncthreads();
    }
    valid[t] = (t < oc) ? 1 : 0;
    if (t == 0) *ocptr = oc;
}

// new = sel @ X / max(count,1)   (K=256, skip zero rows)
__global__ __launch_bounds__(256)
void k_apply(const float* __restrict__ sel, const float* __restrict__ X,
             float* __restrict__ out)
{
    __shared__ float srow[256];
    __shared__ float sRed[256];
    const int a = blockIdx.x, t = threadIdx.x;
    float sv = sel[a * 256 + t];
    srow[t] = sv;
    sRed[t] = sv;
    __syncthreads();
    for (int sft = 128; sft > 0; sft >>= 1) {
        if (t < sft) sRed[t] += sRed[t + sft];
        __syncthreads();
    }
    const float inv = 1.f / fmaxf(sRed[0], 1.f);
    float acc[16];
#pragma unroll
    for (int k = 0; k < 16; ++k) acc[k] = 0.f;
    for (int m = 0; m < 256; ++m) {
        if (srow[m] != 0.f) {
#pragma unroll
            for (int k = 0; k < 16; ++k)
                acc[k] += X[(size_t)m * 4096 + t + 256 * k];
        }
    }
#pragma unroll
    for (int k = 0; k < 16; ++k)
        out[(size_t)a * 4096 + t + 256 * k] = acc[k] * inv;
}

// Bilinear 4x upsample + argmax over valid channels -> index map (as float).
__global__ __launch_bounds__(256)
void k_argmax(const float* __restrict__ attns, const int* __restrict__ ocptr,
              float* __restrict__ out)
{
    const int u = blockIdx.x, v = threadIdx.x;
    const int oc = *ocptr;
    float fy = 63.f * (float)u / 255.f;
    float fx = 63.f * (float)v / 255.f;
    int y0 = (int)fy; if (y0 > 62) y0 = 62;
    int x0 = (int)fx; if (x0 > 62) x0 = 62;
    float wy = fy - (float)y0, wx = fx - (float)x0;
    float best = -INFINITY; int bi = 0;
    for (int c = 0; c < 256; ++c) {
        if (c >= oc) break;                       // invalid -> -inf, never wins
        const float* mp = attns + (size_t)c * 4096;
        float a = mp[y0 * 64 + x0],       b = mp[y0 * 64 + x0 + 1];
        float cc = mp[(y0 + 1) * 64 + x0], d = mp[(y0 + 1) * 64 + x0 + 1];
        float r0 = a + (cc - a) * wy;
        float r1 = b + (d - b) * wy;
        float val = r0 + (r1 - r0) * wx;
        if (val > best) { best = val; bi = c; }   // first-max tie rule
    }
    out[u * 256 + v] = (float)bi;
}

extern "C" void kernel_launch(void* const* d_in, const int* in_sizes, int n_in,
                              void* d_out, int out_size, void* d_ws, size_t ws_size,
                              hipStream_t stream) {
    const float* w64 = (const float*)d_in[0];   // (1,8,4096,4096)
    const float* w32 = (const float*)d_in[1];   // (1,8,1024,1024)
    const float* w16 = (const float*)d_in[2];   // (1,8,256,256)
    const float* w8  = (const float*)d_in[3];   // (1,8,64,64)
    float* out = (float*)d_out;                 // (1,256,256) indices as float

    char* ws = (char*)d_ws;
    size_t off = 0;
    auto alloc = [&](size_t bytes) -> char* {
        char* p = ws + off;
        off = (off + bytes + 255) & ~(size_t)255;
        return p;
    };
    _Float16* Yh    = (_Float16*)alloc((size_t)4096 * 4096 * 2);  // 32 MB
    _Float16* lY    = (_Float16*)alloc((size_t)4096 * 4096 * 2);  // 32 MB (reused as YhT)
    float*    sy    = (float*)   alloc(4096 * 4);
    _Float16* klnum = (_Float16*)alloc((size_t)256 * 4096 * 2);   // 2 MB
    float*    cnt   = (float*)   alloc(256 * 4);
    float*    attA  = (float*)   alloc((size_t)256 * 4096 * 4);   // 4 MB
    float*    attB  = (float*)   alloc((size_t)256 * 4096 * 4);   // 4 MB
    float*    lX    = (float*)   alloc((size_t)256 * 4096 * 4);   // 4 MB
    float*    sx    = (float*)   alloc(256 * 4);
    float*    klm   = (float*)   alloc(256 * 256 * 4);
    float*    sel   = (float*)   alloc(256 * 256 * 4);
    int*      valid = (int*)     alloc(256 * 4);
    int*      ocnt  = (int*)     alloc(4);
    _Float16* YhT   = lY;   // lY region reused after k_cross_kl completes

    // Stage 1: aggregate multi-scale attention -> fp16 Y / log Y / entropies
    k_aggregate<<<4096, 256, 0, stream>>>(w64, w32, w16, w8, Yh, lY, sy);

    // Stage 2: anchor-vs-all KL via WMMA -> binary selection matrix + counts
    k_cross_kl<<<512, 256, 0, stream>>>(Yh, lY, sy, klnum);
    k_rowcount<<<256, 256, 0, stream>>>(klnum, cnt);
    k_transpose_h<<<dim3(128, 128), 256, 0, stream>>>(Yh, YhT);
    k_newattns<<<512, 256, 0, stream>>>(klnum, YhT, cnt, attA);

    // Stage 3: two greedy merge rounds
    k_init_valid<<<1, 256, 0, stream>>>(valid);

    k_prep  <<<256, 256, 0, stream>>>(attA, valid, lX, sx);
    k_klmat <<<256, 256, 0, stream>>>(attA, lX, sx, valid, klm);
    k_greedy<<<1,   256, 0, stream>>>(klm, valid, sel, ocnt);
    k_apply <<<256, 256, 0, stream>>>(sel, attA, attB);

    k_prep  <<<256, 256, 0, stream>>>(attB, valid, lX, sx);
    k_klmat <<<256, 256, 0, stream>>>(attB, lX, sx, valid, klm);
    k_greedy<<<1,   256, 0, stream>>>(klm, valid, sel, ocnt);
    k_apply <<<256, 256, 0, stream>>>(sel, attB, attA);

    // Stage 4: 4x upsample + argmax -> segment index map
    k_argmax<<<256, 256, 0, stream>>>(attA, ocnt, out);
}
